// AttLIF_37692632989801
// MI455X (gfx1250) — matrix operand order repaired and checked by reading, not compiled, not run on hardware
//
#include <hip/hip_runtime.h>
#include <math.h>

// Problem dims (fixed by the reference)
#define TT   64
#define BB   256
#define CIN  1024
#define COUT 1024
#define MM   (TT * BB)   // 16384 rows of the GEMM
#define KK   CIN
#define NN   COUT
#define NSTEP (KK / 32)  // 32 k-steps

// LDS K-row stride in bf16 elements: pad 32 -> 40 (80 B rows).
// Fragment reads: 16 lanes at stride 20 dwords -> 16 disjoint 4-dword bank
// groups (conflict-free), and every b128 access stays 16-byte aligned.
#define LDSK 40
#define SEG  (128 * LDSK)      // one tile in ushort elements (10240 B)
#define SEGB (SEG * 2)         // one tile in bytes

typedef __attribute__((ext_vector_type(16))) __bf16 bf16x16;
typedef __attribute__((ext_vector_type(8)))  float  f32x8;

union Frag {
  bf16x16 v;
  uint4   q[2];
};

union Pk2 {
  __bf16       b[2];
  unsigned int u;
};

// Split two fp32 into packed bf16 hi-pair and lo-residual-pair.
__device__ __forceinline__ void split_pair(float a, float b,
                                           unsigned int& hiu, unsigned int& lou) {
  __bf16 ha = (__bf16)a, hb = (__bf16)b;       // RNE to bf16
  float  ra = a - (float)ha, rb = b - (float)hb;
  Pk2 ph; ph.b[0] = ha;          ph.b[1] = hb;          hiu = ph.u;
  Pk2 pl; pl.b[0] = (__bf16)ra;  pl.b[1] = (__bf16)rb;  lou = pl.u;
}

// ---------------------------------------------------------------------------
// Prepass: split fp32 array into bf16 hi / bf16 lo-residual arrays.
// ---------------------------------------------------------------------------
__global__ __launch_bounds__(256) void split_bf16_kernel(
    const float* __restrict__ src, unsigned short* __restrict__ hi,
    unsigned short* __restrict__ lo, int n4) {
  int i = blockIdx.x * blockDim.x + threadIdx.x;
  if (i >= n4) return;
  float4 v = ((const float4*)src)[i];
  uint2 h, l;
  split_pair(v.x, v.y, h.x, l.x);
  split_pair(v.z, v.w, h.y, l.y);
  ((uint2*)hi)[i] = h;
  ((uint2*)lo)[i] = l;
}

// ---------------------------------------------------------------------------
// Async-fill GEMM: y[m,n] = sum_k x[m,k]*W[n,k] + b[n], inputs pre-split bf16.
// Block tile 128x128, K-step 32, 8 waves (4M x 2N), wave tile 32x64 (2x4 WMMA
// subtiles, 3 bf16 WMMAs each for fp32-accurate split accumulation).
// LDS is double-buffered; tiles are filled with global_load_async_to_lds_b128
// (ASYNCcnt) so the next K-step streams in underneath the current WMMAs.
// ---------------------------------------------------------------------------
__device__ __forceinline__ void issue_stage(
    unsigned ldsb,                                      // stage base byte addr
    const unsigned short* xhi, const unsigned short* xlo,
    const unsigned short* whi, const unsigned short* wlo,
    int tid, int blockM, int blockN, int k0) {
#pragma unroll
  for (int i = 0; i < 2; ++i) {
    int slot = tid + i * 256;            // 0..511 (row, 8-elem chunk) slots
    int row  = slot >> 2;                // 0..127
    int ch   = (slot & 3) << 3;          // elem col: 0,8,16,24
    unsigned goffA = (unsigned)(((blockM + row) * KK + k0 + ch) * 2);
    unsigned goffB = (unsigned)(((blockN + row) * KK + k0 + ch) * 2);
    unsigned lA = ldsb + (unsigned)((row * LDSK + ch) * 2);
    asm volatile("global_load_async_to_lds_b128 %0, %1, %2"
                 :: "v"(lA),            "v"(goffA), "s"((unsigned long long)xhi) : "memory");
    asm volatile("global_load_async_to_lds_b128 %0, %1, %2"
                 :: "v"(lA + SEGB),     "v"(goffA), "s"((unsigned long long)xlo) : "memory");
    asm volatile("global_load_async_to_lds_b128 %0, %1, %2"
                 :: "v"(lA + 2 * SEGB), "v"(goffB), "s"((unsigned long long)whi) : "memory");
    asm volatile("global_load_async_to_lds_b128 %0, %1, %2"
                 :: "v"(lA + 3 * SEGB), "v"(goffB), "s"((unsigned long long)wlo) : "memory");
  }
}

__global__ __launch_bounds__(256) void plif_gemm_async_kernel(
    const unsigned short* __restrict__ xhi, const unsigned short* __restrict__ xlo,
    const unsigned short* __restrict__ whi, const unsigned short* __restrict__ wlo,
    const float* __restrict__ bias, float* __restrict__ y) {
  __shared__ unsigned short lds[2 * 4 * SEG];   // 2 stages x {Ahi,Alo,Bhi,Blo}

  const int tid    = threadIdx.x;
  const int lane   = tid & 31;
  const int wave   = tid >> 5;
  const int waveM  = wave & 3;
  const int waveN  = wave >> 2;
  const int blockM = blockIdx.y * 128;
  const int blockN = blockIdx.x * 128;
  const int lane15 = lane & 15;
  const int hi16   = (lane & 16) ? 1 : 0;

  const unsigned lds0 = (unsigned)(uintptr_t)(&lds[0]);

  f32x8 acc[2][4];
#pragma unroll
  for (int mi = 0; mi < 2; ++mi)
#pragma unroll
    for (int ni = 0; ni < 4; ++ni)
#pragma unroll
      for (int r = 0; r < 8; ++r) acc[mi][ni][r] = 0.0f;

  issue_stage(lds0, xhi, xlo, whi, wlo, tid, blockM, blockN, 0);

  for (int ks = 0; ks < NSTEP; ++ks) {
    const int cur = ks & 1;
    if (ks + 1 < NSTEP) {
      issue_stage(lds0 + (unsigned)(((ks + 1) & 1) * 4 * SEGB),
                  xhi, xlo, whi, wlo, tid, blockM, blockN, (ks + 1) * 32);
      asm volatile("s_wait_asynccnt 8" ::: "memory");   // stage `cur` resident
    } else {
      asm volatile("s_wait_asynccnt 0" ::: "memory");
    }
    __syncthreads();

    const unsigned short* Ahi = &lds[cur * 4 * SEG];
    const unsigned short* Alo = Ahi + SEG;
    const unsigned short* Bhi = Ahi + 2 * SEG;
    const unsigned short* Blo = Ahi + 3 * SEG;

    Frag ahi[2], alo[2];
#pragma unroll
    for (int mi = 0; mi < 2; ++mi) {
      int row = waveM * 32 + mi * 16 + lane15;
      int kb  = hi16 * 8;
      ahi[mi].q[0] = *(const uint4*)(&Ahi[row * LDSK + kb]);
      ahi[mi].q[1] = *(const uint4*)(&Ahi[row * LDSK + kb + 16]);
      alo[mi].q[0] = *(const uint4*)(&Alo[row * LDSK + kb]);
      alo[mi].q[1] = *(const uint4*)(&Alo[row * LDSK + kb + 16]);
    }
    Frag bhi[4], blo[4];
#pragma unroll
    for (int ni = 0; ni < 4; ++ni) {
      int row = waveN * 64 + ni * 16 + lane15;
      int kb  = hi16 * 8;
      bhi[ni].q[0] = *(const uint4*)(&Bhi[row * LDSK + kb]);
      bhi[ni].q[1] = *(const uint4*)(&Bhi[row * LDSK + kb + 16]);
      blo[ni].q[0] = *(const uint4*)(&Blo[row * LDSK + kb]);
      blo[ni].q[1] = *(const uint4*)(&Blo[row * LDSK + kb + 16]);
    }

#pragma unroll
    for (int mi = 0; mi < 2; ++mi)
#pragma unroll
      for (int ni = 0; ni < 4; ++ni) {
        acc[mi][ni] = __builtin_amdgcn_wmma_f32_16x16x32_bf16(
            false, alo[mi].v, false, bhi[ni].v, (short)0, acc[mi][ni], false, false);
        acc[mi][ni] = __builtin_amdgcn_wmma_f32_16x16x32_bf16(
            false, ahi[mi].v, false, blo[ni].v, (short)0, acc[mi][ni], false, false);
        acc[mi][ni] = __builtin_amdgcn_wmma_f32_16x16x32_bf16(
            false, ahi[mi].v, false, bhi[ni].v, (short)0, acc[mi][ni], false, false);
      }
    __syncthreads();   // frag reads done before next stage overwrites buffer
  }

#pragma unroll
  for (int ni = 0; ni < 4; ++ni) {
    int n = blockN + waveN * 64 + ni * 16 + lane15;
    float bv = bias[n];
#pragma unroll
    for (int mi = 0; mi < 2; ++mi) {
      int mbase = blockM + waveM * 32 + mi * 16 + hi16 * 8;
#pragma unroll
      for (int r = 0; r < 8; ++r)
        y[(size_t)(mbase + r) * NN + n] = acc[mi][ni][r] + bv;
    }
  }
}

// ---------------------------------------------------------------------------
// Fallback GEMM (no workspace): convert fp32->bf16 hi/lo in the cooperative
// loader, store packed bf16 to LDS, consume with ds_load_b128 fragments.
// ---------------------------------------------------------------------------
__global__ __launch_bounds__(256) void plif_gemm_cvt_kernel(
    const float* __restrict__ x, const float* __restrict__ W,
    const float* __restrict__ bias, float* __restrict__ y) {
  __shared__ unsigned short Ahi[SEG];
  __shared__ unsigned short Alo[SEG];
  __shared__ unsigned short Bhi[SEG];
  __shared__ unsigned short Blo[SEG];

  const int tid    = threadIdx.x;
  const int lane   = tid & 31;
  const int wave   = tid >> 5;
  const int waveM  = wave & 3;
  const int waveN  = wave >> 2;
  const int blockM = blockIdx.y * 128;
  const int blockN = blockIdx.x * 128;
  const int lane15 = lane & 15;
  const int hi16   = (lane & 16) ? 1 : 0;

  f32x8 acc[2][4];
#pragma unroll
  for (int mi = 0; mi < 2; ++mi)
#pragma unroll
    for (int ni = 0; ni < 4; ++ni)
#pragma unroll
      for (int r = 0; r < 8; ++r) acc[mi][ni][r] = 0.0f;

  for (int k0 = 0; k0 < KK; k0 += 32) {
#pragma unroll
    for (int i = 0; i < 4; ++i) {
      int slot = tid + i * 256;
      int row  = slot >> 3;
      int c4   = (slot & 7) << 2;
      float4 av = *(const float4*)(x + (size_t)(blockM + row) * KK + (k0 + c4));
      uint2 h, l;
      split_pair(av.x, av.y, h.x, l.x);
      split_pair(av.z, av.w, h.y, l.y);
      *(uint2*)(&Ahi[row * LDSK + c4]) = h;
      *(uint2*)(&Alo[row * LDSK + c4]) = l;
      float4 bv = *(const float4*)(W + (size_t)(blockN + row) * KK + (k0 + c4));
      split_pair(bv.x, bv.y, h.x, l.x);
      split_pair(bv.z, bv.w, h.y, l.y);
      *(uint2*)(&Bhi[row * LDSK + c4]) = h;
      *(uint2*)(&Blo[row * LDSK + c4]) = l;
    }
    __syncthreads();

    Frag ahi[2], alo[2];
#pragma unroll
    for (int mi = 0; mi < 2; ++mi) {
      int row = waveM * 32 + mi * 16 + lane15;
      int kb  = hi16 * 8;
      ahi[mi].q[0] = *(const uint4*)(&Ahi[row * LDSK + kb]);
      ahi[mi].q[1] = *(const uint4*)(&Ahi[row * LDSK + kb + 16]);
      alo[mi].q[0] = *(const uint4*)(&Alo[row * LDSK + kb]);
      alo[mi].q[1] = *(const uint4*)(&Alo[row * LDSK + kb + 16]);
    }
    Frag bhi[4], blo[4];
#pragma unroll
    for (int ni = 0; ni < 4; ++ni) {
      int row = waveN * 64 + ni * 16 + lane15;
      int kb  = hi16 * 8;
      bhi[ni].q[0] = *(const uint4*)(&Bhi[row * LDSK + kb]);
      bhi[ni].q[1] = *(const uint4*)(&Bhi[row * LDSK + kb + 16]);
      blo[ni].q[0] = *(const uint4*)(&Blo[row * LDSK + kb]);
      blo[ni].q[1] = *(const uint4*)(&Blo[row * LDSK + kb + 16]);
    }

#pragma unroll
    for (int mi = 0; mi < 2; ++mi)
#pragma unroll
      for (int ni = 0; ni < 4; ++ni) {
        acc[mi][ni] = __builtin_amdgcn_wmma_f32_16x16x32_bf16(
            false, alo[mi].v, false, bhi[ni].v, (short)0, acc[mi][ni], false, false);
        acc[mi][ni] = __builtin_amdgcn_wmma_f32_16x16x32_bf16(
            false, ahi[mi].v, false, blo[ni].v, (short)0, acc[mi][ni], false, false);
        acc[mi][ni] = __builtin_amdgcn_wmma_f32_16x16x32_bf16(
            false, ahi[mi].v, false, bhi[ni].v, (short)0, acc[mi][ni], false, false);
      }
    __syncthreads();
  }

#pragma unroll
  for (int ni = 0; ni < 4; ++ni) {
    int n = blockN + waveN * 64 + ni * 16 + lane15;
    float bv = bias[n];
#pragma unroll
    for (int mi = 0; mi < 2; ++mi) {
      int mbase = blockM + waveM * 32 + mi * 16 + hi16 * 8;
#pragma unroll
      for (int r = 0; r < 8; ++r)
        y[(size_t)(mbase + r) * NN + n] = acc[mi][ni][r] + bv;
    }
  }
}

// ---------------------------------------------------------------------------
// PLIF scan, in place on d_out (y -> spikes).
//   v += (y - v) * sigmoid(w);  s = (v >= 1);  v *= (1 - s)
// ---------------------------------------------------------------------------
__global__ __launch_bounds__(256) void plif_scan_kernel(
    float* __restrict__ yspk, const float* __restrict__ wptr) {
  const float decay = 1.0f / (1.0f + __expf(-wptr[0]));
  const size_t C4 = (size_t)BB * COUT / 4;
  size_t c4 = (size_t)blockIdx.x * blockDim.x + threadIdx.x;
  float4* p = (float4*)yspk;
  float4 v = make_float4(0.f, 0.f, 0.f, 0.f);
#pragma unroll 1
  for (int t = 0; t < TT; ++t) {
    size_t off = (size_t)t * C4 + c4;
    float4 yv = p[off];
    float4 s;
    v.x += (yv.x - v.x) * decay;  s.x = (v.x >= 1.0f) ? 1.0f : 0.0f;  v.x *= (1.0f - s.x);
    v.y += (yv.y - v.y) * decay;  s.y = (v.y >= 1.0f) ? 1.0f : 0.0f;  v.y *= (1.0f - s.y);
    v.z += (yv.z - v.z) * decay;  s.z = (v.z >= 1.0f) ? 1.0f : 0.0f;  v.z *= (1.0f - s.z);
    v.w += (yv.w - v.w) * decay;  s.w = (v.w >= 1.0f) ? 1.0f : 0.0f;  v.w *= (1.0f - s.w);
    p[off] = s;
  }
}

extern "C" void kernel_launch(void* const* d_in, const int* in_sizes, int n_in,
                              void* d_out, int out_size, void* d_ws, size_t ws_size,
                              hipStream_t stream) {
  (void)in_sizes; (void)n_in; (void)out_size;
  const float* x = (const float*)d_in[0];
  const float* W = (const float*)d_in[1];
  const float* b = (const float*)d_in[2];
  const float* w = (const float*)d_in[3];
  float* out = (float*)d_out;

  const size_t xh = (size_t)MM * KK * 2;   // 33,554,432 B per x half
  const size_t wh = (size_t)NN * KK * 2;   //  2,097,152 B per W half
  const size_t need = 2 * xh + 2 * wh;

  dim3 grid(NN / 128, MM / 128);           // 8 x 128 workgroups

  if (ws_size >= need) {
    unsigned short* xhi = (unsigned short*)d_ws;
    unsigned short* xlo = (unsigned short*)((char*)d_ws + xh);
    unsigned short* whi = (unsigned short*)((char*)d_ws + 2 * xh);
    unsigned short* wlo = (unsigned short*)((char*)d_ws + 2 * xh + wh);
    int xn4 = MM * KK / 4, wn4 = NN * KK / 4;
    split_bf16_kernel<<<xn4 / 256, 256, 0, stream>>>(x, xhi, xlo, xn4);
    split_bf16_kernel<<<wn4 / 256, 256, 0, stream>>>(W, whi, wlo, wn4);
    plif_gemm_async_kernel<<<grid, 256, 0, stream>>>(xhi, xlo, whi, wlo, b, out);
  } else {
    plif_gemm_cvt_kernel<<<grid, 256, 0, stream>>>(x, W, b, out);
  }

  int scan_threads = (BB * COUT) / 4;      // 65536
  plif_scan_kernel<<<scan_threads / 256, 256, 0, stream>>>(out, w);
}
